// BanditLayer_4329327035170
// MI455X (gfx1250) — compile-verified, matrix operand order checked by reading, not run on hardware
//
#include <hip/hip_runtime.h>
#include <stdint.h>

// ---------------------------------------------------------------------------
// BanditLayer inference: C = X (M x P) * W^T + bias, fp32 I/O.
//
// Pass 1 (once per launch): split X and W into bf16 hi/lo planes in d_ws
//   (x = hi + lo, lo = bf16(x - float(hi))) at HBM streaming rate.
// Pass 2: bf16x3 split-precision GEMM on V_WMMA_F32_16X16X32_BF16:
//   C ~= Xhi*Whi^T + Xhi*Wlo^T + Xlo*Whi^T  (fp32-class accuracy).
//   Tiles staged with GLOBAL_LOAD_ASYNC_TO_LDS_B128, double-buffered and
//   pipelined on ASYNCcnt; main loop contains only ds_load_b128 + WMMA.
// Fallback (ws too small): single-kernel variant converting per-fragment.
// ---------------------------------------------------------------------------

typedef __attribute__((ext_vector_type(16))) __bf16 bf16x16;
typedef __attribute__((ext_vector_type(8)))  __bf16 bf16x8;
typedef __attribute__((ext_vector_type(4)))  __bf16 bf16x4;
typedef __attribute__((ext_vector_type(8)))  float  f32x8;
typedef __attribute__((ext_vector_type(4)))  float  f32x4;
typedef __attribute__((ext_vector_type(4)))  int    i32x4;

#define TM 128
#define TN 128
#define TK 32
#define BSTR 40                    // bf16 LDS row stride (32 + 8 pad -> 80B)
#define TILE_H (128 * BSTR)        // bf16 elements per tile (10240 B)
#define FSTR 36                    // fp32 LDS row stride (fallback kernel)
#define FTILE (128 * FSTR)

#if __has_builtin(__builtin_amdgcn_global_load_async_to_lds_b128) && \
    __has_builtin(__builtin_amdgcn_s_wait_asynccnt)
#define USE_ASYNC_LDS 1
#else
#define USE_ASYNC_LDS 0
#endif

// ---------------------------------------------------------------------------
// Pass 1: fp32 -> bf16 hi/lo planes.
// ---------------------------------------------------------------------------
__global__ void __launch_bounds__(256)
split_planes(const float* __restrict__ src, uint16_t* __restrict__ hi_u,
             uint16_t* __restrict__ lo_u, int n4)
{
    int i = blockIdx.x * 256 + threadIdx.x;
    if (i >= n4) return;
    f32x4 v = ((const f32x4*)src)[i];
    bf16x4 h, l;
#pragma unroll
    for (int j = 0; j < 4; ++j) {
        __bf16 hj = (__bf16)v[j];
        h[j] = hj;
        l[j] = (__bf16)(v[j] - (float)hj);   // exact residual
    }
    ((bf16x4*)hi_u)[i] = h;
    ((bf16x4*)lo_u)[i] = l;
}

// ---------------------------------------------------------------------------
// Shared helpers.
// ---------------------------------------------------------------------------

// Combine the two 8-element chunks of a bf16 fragment row into a v16 operand.
// Per ISA 7.12.2 (16-bit A/B 16x32): lanes 0-15 hold K 0..7 & 16..23,
// lanes 16-31 hold K 8..15 & 24..31, sequential within VGPRs.
__device__ __forceinline__ bf16x16 load_frag(const __bf16* row, int khalf) {
    bf16x8 a = *(const bf16x8*)(row + khalf);
    bf16x8 b = *(const bf16x8*)(row + 16 + khalf);
    return __builtin_shufflevector(a, b, 0, 1, 2, 3, 4, 5, 6, 7,
                                         8, 9, 10, 11, 12, 13, 14, 15);
}

__device__ __forceinline__ void async_b128(const void* g, void* l) {
#if USE_ASYNC_LDS
    __builtin_amdgcn_global_load_async_to_lds_b128(
        (i32x4*)(uintptr_t)g, (i32x4*)l, 0, 0);
#else
    *(i32x4*)l = *(const i32x4*)g;
#endif
}

// Map a linear block id so groups of 8 M-tiles sweep all N: the W panel
// stays L2-resident (192MB L2) while X panels rotate.
__device__ __forceinline__ void swizzle_bxy(int bid, int nbx, int nby,
                                            int& bx, int& by) {
    const int GM = 8;
    int band  = GM * nbx;
    int group = bid / band;
    int rem   = bid - group * band;
    int gsz   = nby - group * GM; if (gsz > GM) gsz = GM;
    by = group * GM + (rem % gsz);
    bx = rem / gsz;
}

// ---------------------------------------------------------------------------
// Pass 2: bf16x3 WMMA GEMM over pre-split planes.
// LDS: [2 buffers][4 tiles: Ahi, Alo, Bhi, Blo][128][BSTR] bf16 = 80 KB.
// ---------------------------------------------------------------------------

// Stage one k-step: 4 bf16 tiles of 128x32. 256 threads x 2 x b128 per tile
// = 8 async ops per thread per stage.
__device__ __forceinline__ void stage4(const __bf16* a_hi, const __bf16* a_lo,
                                       const __bf16* b_hi, const __bf16* b_lo,
                                       __bf16* sbuf, int tid, int K)
{
#pragma unroll
    for (int i = 0; i < 2; ++i) {
        int idx = i * 256 + tid;
        int row = idx >> 2;             // 0..127
        int seg = (idx & 3) << 3;       // bf16 offset in row: 0,8,16,24
        size_t goff = (size_t)row * K + seg;
        int    loff = row * BSTR + seg;
        async_b128(a_hi + goff, sbuf + 0 * TILE_H + loff);
        async_b128(a_lo + goff, sbuf + 1 * TILE_H + loff);
        async_b128(b_hi + goff, sbuf + 2 * TILE_H + loff);
        async_b128(b_lo + goff, sbuf + 3 * TILE_H + loff);
    }
}

__global__ void __launch_bounds__(256)
bandit_gemm_planes(const uint16_t* __restrict__ Xhi_u,
                   const uint16_t* __restrict__ Xlo_u,
                   const uint16_t* __restrict__ Whi_u,
                   const uint16_t* __restrict__ Wlo_u,
                   const float* __restrict__ bias,
                   float* __restrict__ out,
                   int M, int N, int K)
{
    extern __shared__ char dynsmem[];
    __bf16* smem = (__bf16*)dynsmem;

    const __bf16* Xhi = (const __bf16*)Xhi_u;
    const __bf16* Xlo = (const __bf16*)Xlo_u;
    const __bf16* Whi = (const __bf16*)Whi_u;
    const __bf16* Wlo = (const __bf16*)Wlo_u;

    const int tid   = threadIdx.x;
    const int lane  = tid & 31;
    const int wave  = tid >> 5;
    const int wm    = wave >> 2;            // 0..1 (64 M-rows each)
    const int wn    = wave & 3;             // 0..3 (32 N-cols each)
    const int lrow  = lane & 15;
    const int khalf = (lane >> 4) << 3;     // 0 or 8

    int bx, by;
    swizzle_bxy(blockIdx.x, N / TN, M / TM, bx, by);
    const int m0 = by * TM;
    const int n0 = bx * TN;

    const __bf16* aH = Xhi + (size_t)m0 * K;
    const __bf16* aL = Xlo + (size_t)m0 * K;
    const __bf16* bH = Whi + (size_t)n0 * K;
    const __bf16* bL = Wlo + (size_t)n0 * K;

    f32x8 acc[4][2] = {};
    const int nk = K / TK;

    // Prologue: stage k-step 0 into buffer 0.
    stage4(aH, aL, bH, bL, smem, tid, K);

    for (int kt = 0; kt < nk; ++kt) {
        const int cur = kt & 1;

        if (kt + 1 < nk) {
            // Issue next tile batch, then counted-wait for only the current
            // batch (async loads complete in order).
            int ko = (kt + 1) * TK;
            stage4(aH + ko, aL + ko, bH + ko, bL + ko,
                   smem + (cur ^ 1) * 4 * TILE_H, tid, K);
#if USE_ASYNC_LDS
            __builtin_amdgcn_s_wait_asynccnt(8);
#endif
        } else {
#if USE_ASYNC_LDS
            __builtin_amdgcn_s_wait_asynccnt(0);
#endif
        }
        __syncthreads();

        const __bf16* tAh = smem + cur * 4 * TILE_H;
        const __bf16* tAl = tAh + TILE_H;
        const __bf16* tBh = tAl + TILE_H;
        const __bf16* tBl = tBh + TILE_H;

        bf16x16 bh[2], bl[2];
#pragma unroll
        for (int fn = 0; fn < 2; ++fn) {
            int r = (wn * 32 + fn * 16 + lrow) * BSTR;
            bh[fn] = load_frag(tBh + r, khalf);
            bl[fn] = load_frag(tBl + r, khalf);
        }

#pragma unroll
        for (int fm = 0; fm < 4; ++fm) {
            int r = (wm * 64 + fm * 16 + lrow) * BSTR;
            bf16x16 ah = load_frag(tAh + r, khalf);
            bf16x16 al = load_frag(tAl + r, khalf);
#pragma unroll
            for (int fn = 0; fn < 2; ++fn) {
                acc[fm][fn] = __builtin_amdgcn_wmma_f32_16x16x32_bf16(
                    false, ah, false, bh[fn], (short)0, acc[fm][fn], false, false);
                acc[fm][fn] = __builtin_amdgcn_wmma_f32_16x16x32_bf16(
                    false, ah, false, bl[fn], (short)0, acc[fm][fn], false, false);
                acc[fm][fn] = __builtin_amdgcn_wmma_f32_16x16x32_bf16(
                    false, al, false, bh[fn], (short)0, acc[fm][fn], false, false);
            }
        }
        __syncthreads();
    }

    // Epilogue: bias add + fp32 store.
    // D layout: element (vgpr g, lane) -> m = g + 8*(lane>=16), n = lane&15.
#pragma unroll
    for (int fn = 0; fn < 2; ++fn) {
        int n    = n0 + wn * 32 + fn * 16 + lrow;
        float bv = bias[n];
#pragma unroll
        for (int fm = 0; fm < 4; ++fm) {
            int mbase = m0 + wm * 64 + fm * 16 + ((lane >> 4) << 3);
#pragma unroll
            for (int g = 0; g < 8; ++g) {
                out[(size_t)(mbase + g) * N + n] = acc[fm][fn][g] + bv;
            }
        }
    }
}

// ---------------------------------------------------------------------------
// Fallback (ws too small): on-the-fly per-fragment split (round-3 variant).
// ---------------------------------------------------------------------------
__device__ __forceinline__ void cvt_frag(const float* rowp, int khalf,
                                         bf16x16& h, bf16x16& l)
{
    f32x8 c0 = *(const f32x8*)(rowp + khalf);
    f32x8 c1 = *(const f32x8*)(rowp + 16 + khalf);
#pragma unroll
    for (int j = 0; j < 8; ++j) {
        __bf16 h0 = (__bf16)c0[j];
        h[j]     = h0;
        l[j]     = (__bf16)(c0[j] - (float)h0);
        __bf16 h1 = (__bf16)c1[j];
        h[8 + j] = h1;
        l[8 + j] = (__bf16)(c1[j] - (float)h1);
    }
}

__global__ void __launch_bounds__(256)
bandit_gemm_inline(const float* __restrict__ X, const float* __restrict__ W,
                   const float* __restrict__ bias, float* __restrict__ out,
                   int M, int N, int K)
{
    extern __shared__ char dynsmem[];
    float* sA = (float*)dynsmem;         // [2][FTILE]
    float* sB = sA + 2 * FTILE;          // [2][FTILE]

    const int tid   = threadIdx.x;
    const int lane  = tid & 31;
    const int wave  = tid >> 5;
    const int wm    = wave >> 2;
    const int wn    = wave & 3;
    const int lrow  = lane & 15;
    const int khalf = (lane >> 4) << 3;

    int bx, by;
    swizzle_bxy(blockIdx.x, N / TN, M / TM, bx, by);
    const int m0 = by * TM;
    const int n0 = bx * TN;

    const float* gA0 = X + (size_t)m0 * K;
    const float* gB0 = W + (size_t)n0 * K;

    f32x8 acc[4][2] = {};
    const int nk = K / TK;

    auto stage = [&](const float* gA, const float* gB, float* dA, float* dB) {
#pragma unroll
        for (int i = 0; i < 4; ++i) {
            int idx = i * 256 + tid;
            int row = idx >> 3;
            int seg = (idx & 7) << 2;
            async_b128(gA + (size_t)row * K + seg, dA + row * FSTR + seg);
            async_b128(gB + (size_t)row * K + seg, dB + row * FSTR + seg);
        }
    };

    stage(gA0, gB0, sA, sB);

    for (int kt = 0; kt < nk; ++kt) {
        const int cur = kt & 1;
        if (kt + 1 < nk) {
            stage(gA0 + (kt + 1) * TK, gB0 + (kt + 1) * TK,
                  sA + (cur ^ 1) * FTILE, sB + (cur ^ 1) * FTILE);
#if USE_ASYNC_LDS
            __builtin_amdgcn_s_wait_asynccnt(8);
#endif
        } else {
#if USE_ASYNC_LDS
            __builtin_amdgcn_s_wait_asynccnt(0);
#endif
        }
        __syncthreads();

        const float* bufA = sA + cur * FTILE;
        const float* bufB = sB + cur * FTILE;

        bf16x16 bh[2], bl[2];
#pragma unroll
        for (int fn = 0; fn < 2; ++fn)
            cvt_frag(bufB + (wn * 32 + fn * 16 + lrow) * FSTR, khalf,
                     bh[fn], bl[fn]);
#pragma unroll
        for (int fm = 0; fm < 4; ++fm) {
            bf16x16 ah, al;
            cvt_frag(bufA + (wm * 64 + fm * 16 + lrow) * FSTR, khalf, ah, al);
#pragma unroll
            for (int fn = 0; fn < 2; ++fn) {
                acc[fm][fn] = __builtin_amdgcn_wmma_f32_16x16x32_bf16(
                    false, ah, false, bh[fn], (short)0, acc[fm][fn], false, false);
                acc[fm][fn] = __builtin_amdgcn_wmma_f32_16x16x32_bf16(
                    false, ah, false, bl[fn], (short)0, acc[fm][fn], false, false);
                acc[fm][fn] = __builtin_amdgcn_wmma_f32_16x16x32_bf16(
                    false, al, false, bh[fn], (short)0, acc[fm][fn], false, false);
            }
        }
        __syncthreads();
    }

#pragma unroll
    for (int fn = 0; fn < 2; ++fn) {
        int n    = n0 + wn * 32 + fn * 16 + lrow;
        float bv = bias[n];
#pragma unroll
        for (int fm = 0; fm < 4; ++fm) {
            int mbase = m0 + wm * 64 + fm * 16 + ((lane >> 4) << 3);
#pragma unroll
            for (int g = 0; g < 8; ++g)
                out[(size_t)(mbase + g) * N + n] = acc[fm][fn][g] + bv;
        }
    }
}

// ---------------------------------------------------------------------------
extern "C" void kernel_launch(void* const* d_in, const int* in_sizes, int n_in,
                              void* d_out, int out_size, void* d_ws, size_t ws_size,
                              hipStream_t stream) {
    const float* X    = (const float*)d_in[0];   // (M, P) fp32
    const float* W    = (const float*)d_in[1];   // (N, P) fp32
    const float* bias = (const float*)d_in[2];   // (N,)   fp32
    float* out        = (float*)d_out;           // (M, N) fp32

    const int N = in_sizes[2];                   // 8192
    const int P = in_sizes[1] / N;               // 4096
    const int M = in_sizes[0] / P;               // 4096

    const int nblocks = (M / TM) * (N / TN);

    const size_t nA = (size_t)M * P;             // X elements
    const size_t nB = (size_t)N * P;             // W elements
    const size_t need = 2 * (nA + nB) * sizeof(uint16_t);  // hi+lo planes

    if (ws_size >= need) {
        uint16_t* xhi = (uint16_t*)d_ws;
        uint16_t* xlo = xhi + nA;
        uint16_t* whi = xlo + nA;
        uint16_t* wlo = whi + nB;

        int a4 = (int)(nA / 4), b4 = (int)(nB / 4);
        split_planes<<<(a4 + 255) / 256, 256, 0, stream>>>(X, xhi, xlo, a4);
        split_planes<<<(b4 + 255) / 256, 256, 0, stream>>>(W, whi, wlo, b4);

        const size_t lds_bytes = 2 * 4 * (size_t)TILE_H * sizeof(uint16_t); // 81920
        bandit_gemm_planes<<<nblocks, 256, lds_bytes, stream>>>(
            xhi, xlo, whi, wlo, bias, out, M, N, P);
    } else {
        const size_t lds_bytes = 4 * (size_t)FTILE * sizeof(float);         // 73728
        bandit_gemm_inline<<<nblocks, 256, lds_bytes, stream>>>(
            X, W, bias, out, M, N, P);
    }
}
